// DRConv2d_32968168964300
// MI455X (gfx1250) — matrix operand order repaired and checked by reading, not compiled, not run on hardware
//
#include <hip/hip_runtime.h>
#include <hip/hip_bf16.h>

// ---------------- problem constants ----------------
#define R_    8
#define CIN_  64
#define COUT_ 64
#define B_    8
#define H_    126
#define W_    126
#define HO_   124
#define WO_   124
#define KK_   576          // reduction length: k = (kh*3+kw)*64 + ci
#define HT_   16           // 16 h-tiles of 8 rows (last tile: 4 valid rows)

typedef __bf16 bf16_t;
typedef __attribute__((ext_vector_type(16))) __bf16 v16bf;
typedef __attribute__((ext_vector_type(8)))  __bf16 v8bf;
typedef __attribute__((ext_vector_type(8)))  float  v8f;
typedef int v4i_g __attribute__((vector_size(16)));   // matches builtin param type

// ---------------------------------------------------------------------------
// Kernel 1: adaptive avg-pool x -> pooled[B,64,3,3]
// ---------------------------------------------------------------------------
__global__ __launch_bounds__(256) void pool_kernel(const float* __restrict__ x,
                                                   float* __restrict__ pooled) {
  int blk = blockIdx.x;            // b*9 + s
  int b = blk / 9, s = blk % 9;
  int kh = s / 3, kw = s % 3;
  int tid = threadIdx.x;
  int ci = tid >> 2, q = tid & 3;
  const float* xp = x + ((size_t)(b * CIN_ + ci) * H_ + kh * 42) * W_ + kw * 42;
  float sum = 0.f;
  for (int r = q; r < 42; r += 4) {
    const float* row = xp + (size_t)r * W_;
    #pragma unroll 7
    for (int c = 0; c < 42; ++c) sum += row[c];
  }
  __shared__ float red[256];
  red[tid] = sum;
  __syncthreads();
  if (q == 0) {
    float t = red[tid] + red[tid + 1] + red[tid + 2] + red[tid + 3];
    pooled[(b * CIN_ + ci) * 9 + s] = t * (1.0f / 1764.0f);
  }
}

// ---------------------------------------------------------------------------
// Kernel 2: s1 = sigmoid(w1 @ pooled + b1); kern[b,g,co,k] bf16, k = s*64+ci.
// ---------------------------------------------------------------------------
__global__ __launch_bounds__(256) void kerngen_kernel(
    const float* __restrict__ pooled,
    const float* __restrict__ w1, const float* __restrict__ b1,
    const float* __restrict__ w2, const float* __restrict__ b2,
    bf16_t* __restrict__ kern) {
  int blk = blockIdx.x;            // b*8 + g
  int g = blk & 7;
  int b = blk >> 3;
  int tid = threadIdx.x;
  __shared__ float pl[CIN_ * 9];
  __shared__ float s1[8 * 9];      // [i][s]
  for (int t = tid; t < CIN_ * 9; t += 256) pl[t] = pooled[b * CIN_ * 9 + t];
  __syncthreads();
  if (tid < 72) {
    int i = tid / 9, s = tid % 9;
    int o1 = g * 8 + i;
    float acc = b1[o1];
    for (int c = 0; c < CIN_; ++c) acc += pl[c * 9 + s] * w1[o1 * CIN_ + c];
    s1[i * 9 + s] = 1.0f / (1.0f + __expf(-acc));
  }
  __syncthreads();
  for (int o = tid; o < CIN_ * COUT_; o += 256) {  // o = co*64 + ci
    int co = o >> 6, ci = o & 63;
    float wv[8];
    #pragma unroll
    for (int i = 0; i < 8; ++i) wv[i] = w2[((size_t)g * 4096 + o) * 8 + i];
    float bias = b2[g * 4096 + o];
    bf16_t* kp = kern + ((size_t)blk * COUT_ + co) * KK_ + ci;
    #pragma unroll
    for (int s = 0; s < 9; ++s) {
      float acc = bias;
      #pragma unroll
      for (int i = 0; i < 8; ++i) acc += s1[i * 9 + s] * wv[i];
      kp[s * 64] = (bf16_t)acc;
    }
  }
}

// ---------------------------------------------------------------------------
// Kernel 3: transpose+convert x -> xbfT[b][h][w][ci] bf16 (ci contiguous).
// ---------------------------------------------------------------------------
__global__ __launch_bounds__(256) void cvt_kernel(const float* __restrict__ x,
                                                  bf16_t* __restrict__ xbfT) {
  int t = blockIdx.x * 256 + threadIdx.x;          // over B*H*W
  if (t >= B_ * H_ * W_) return;
  int w = t % W_;
  int rest = t / W_;
  int h = rest % H_;
  int b = rest / H_;
  const float* xp = x + ((size_t)b * CIN_ * H_ + h) * W_ + w;
  bf16_t v[CIN_];
  #pragma unroll
  for (int ci = 0; ci < CIN_; ++ci) v[ci] = (bf16_t)xp[(size_t)ci * H_ * W_];
  v8bf* dst = (v8bf*)(xbfT + (size_t)t * CIN_);
  const v8bf* src = (const v8bf*)v;
  #pragma unroll
  for (int k = 0; k < 8; ++k) dst[k] = src[k];
}

// ---------------------------------------------------------------------------
// Kernel 4: guide conv (exact f32) + argmax -> u8 index map.
// ---------------------------------------------------------------------------
__global__ __launch_bounds__(256) void guide_kernel(
    const float* __restrict__ x, const float* __restrict__ wg,
    const float* __restrict__ bg, unsigned char* __restrict__ idxmap) {
  __shared__ float wgl[8 * KK_];                   // [r][ci*9 + s]
  int tid = threadIdx.x;
  for (int t = tid; t < 8 * KK_; t += 256) wgl[t] = wg[t];
  __syncthreads();
  int pix = blockIdx.x * 256 + tid;
  if (pix >= B_ * HO_ * WO_) return;
  int w = pix % WO_;
  int rest = pix / WO_;
  int h = rest % HO_;
  int b = rest / HO_;
  float acc[8];
  #pragma unroll
  for (int r = 0; r < 8; ++r) acc[r] = bg[r];
  const float* xb = x + (size_t)b * CIN_ * H_ * W_;
  for (int ci = 0; ci < CIN_; ++ci) {
    const float* xc = xb + (size_t)ci * H_ * W_ + (size_t)h * W_ + w;
    #pragma unroll
    for (int kh = 0; kh < 3; ++kh)
      #pragma unroll
      for (int kw = 0; kw < 3; ++kw) {
        float xv = xc[kh * W_ + kw];
        int wi = ci * 9 + kh * 3 + kw;
        #pragma unroll
        for (int r = 0; r < 8; ++r) acc[r] += xv * wgl[r * KK_ + wi];
      }
  }
  int best = 0; float bv = acc[0];
  #pragma unroll
  for (int r = 1; r < 8; ++r) if (acc[r] > bv) { bv = acc[r]; best = r; }
  idxmap[pix] = (unsigned char)best;
}

// ---------------------------------------------------------------------------
// Kernel 5: main dynamic conv, WMMA bf16, 8 output rows per block.
// Block = 128 threads (4 waves); wave wv owns cout tile [wv*16, wv*16+16).
// LDS tile: 10 rows x 18 cols x 64 ci bf16 = 23 KB, feeds 8 output rows.
// Staging uses CDNA5 async global->LDS (ASYNCcnt) when the builtin exists.
// ---------------------------------------------------------------------------
__global__ __launch_bounds__(128) void drconv_main(
    const bf16_t* __restrict__ xbfT, const bf16_t* __restrict__ kern,
    const unsigned char* __restrict__ idxmap, float* __restrict__ out) {
  int blk = blockIdx.x;                 // wt + 8*(ht + 16*b)
  int wt = blk & 7;
  int rest = blk >> 3;
  int ht = rest % HT_;
  int b = rest / HT_;
  int h0 = ht * 8;
  int w0 = wt * 16;
  int wvalid = min(16, WO_ - w0);
  int nrows = min(8, HO_ - h0);
  int tid  = threadIdx.x;
  int lane = tid & 31;
  int wv   = tid >> 5;

  __shared__ __attribute__((aligned(16))) bf16_t xs[10 * 18 * 64]; // [row][col][ci]
  __shared__ unsigned char idxt[8 * 16];

  // ---- stage 10x18x64 bf16 patch: 1440 x 16B chunks across 128 threads ----
  for (int c = tid; c < 10 * 18 * 8; c += 128) {
    int ci8 = c & 7;                    // which 16B chunk of the 64-ci run
    int rc  = c >> 3;
    int col = rc % 18;
    int row = rc / 18;
    int gr = h0 + row; if (gr > H_ - 1) gr = H_ - 1;   // clamp (unused rows)
    int gc = w0 + col; if (gc > W_ - 1) gc = W_ - 1;   // clamp (masked cols)
    const bf16_t* gp = xbfT + (((size_t)(b * H_ + gr) * W_ + gc) * CIN_ + ci8 * 8);
    bf16_t* lp = &xs[((row * 18 + col) * 64 + ci8 * 8)];
#if __has_builtin(__builtin_amdgcn_global_load_async_to_lds_b128)
    __builtin_amdgcn_global_load_async_to_lds_b128(
        (__attribute__((address_space(1))) v4i_g*)gp,
        (__attribute__((address_space(3))) v4i_g*)lp, 0, 0);
#else
    *(v8bf*)lp = *(const v8bf*)gp;
#endif
  }
  if (tid < 128) {
    int r = tid >> 4, n = tid & 15;
    int gh = h0 + r, gw = w0 + n;
    idxt[tid] = (gh < HO_ && gw < WO_)
                    ? idxmap[((size_t)b * HO_ + gh) * WO_ + gw]
                    : (unsigned char)255;
  }
#if __has_builtin(__builtin_amdgcn_global_load_async_to_lds_b128)
#if __has_builtin(__builtin_amdgcn_s_wait_asynccnt)
  __builtin_amdgcn_s_wait_asynccnt(0);
#else
  asm volatile("s_wait_asynccnt 0" ::: "memory");
#endif
#endif
  __syncthreads();

  int half = lane >> 4;                 // WMMA lane halves
  int nn   = lane & 15;                 // B/D column (pixel in tile)
  int mrow = lane & 15;                 // A row (cout within 16-tile)
  int cobase = wv * 16;

  for (int r = 0; r < nrows; ++r) {
    // per-row region presence bitmask (identical in all lanes -> scalar)
    unsigned mask = 0;
    #pragma unroll
    for (int n = 0; n < 16; ++n) {
      unsigned id = idxt[r * 16 + n];
      if (id < 8) mask |= 1u << id;
    }
    mask = (unsigned)__builtin_amdgcn_readfirstlane((int)mask);

    for (int g = 0; g < 8; ++g) {
      if (!((mask >> g) & 1u)) continue;       // uniform: skip absent regions
      v8f acc = {};
      const bf16_t* kp =
          kern + ((size_t)((b * 8 + g) * COUT_) + cobase + mrow) * KK_;
      #pragma unroll
      for (int kc = 0; kc < 18; ++kc) {        // K = 576 = 18 x 32
        int kbase = kc * 32;
        // A fragment 16x32 bf16: lane holds k = 16*(j>=8) + 8*half + (j&7)
        v8bf a0 = *(const v8bf*)(kp + kbase + 8 * half);
        v8bf a1 = *(const v8bf*)(kp + kbase + 16 + 8 * half);
        v16bf a = __builtin_shufflevector(a0, a1, 0, 1, 2, 3, 4, 5, 6, 7,
                                          8, 9, 10, 11, 12, 13, 14, 15);
        // B fragment 32x16 bf16: lane column nn, k = kbase + 16*(j>=8) + 8*half + (j&7)
        int s   = kc >> 1;
        int ci0 = (kc & 1) << 5;
        int kh = s / 3, kw = s % 3;
        const bf16_t* bp =
            &xs[(((r + kh) * 18) + (nn + kw)) * 64 + ci0 + 8 * half];
        v8bf b0  = *(const v8bf*)(bp);
        v8bf b1v = *(const v8bf*)(bp + 16);
        v16bf bm = __builtin_shufflevector(b0, b1v, 0, 1, 2, 3, 4, 5, 6, 7,
                                           8, 9, 10, 11, 12, 13, 14, 15);
        acc = __builtin_amdgcn_wmma_f32_16x16x32_bf16(
            false, a, false, bm, (short)0, acc, false, false);
      }
      // masked store: only pixels whose argmax selected region g
      if ((nn < wvalid) && (idxt[r * 16 + nn] == (unsigned char)g)) {
        int wpix = w0 + nn;
        int hpix = h0 + r;
        #pragma unroll
        for (int v = 0; v < 8; ++v) {          // D: row = v + 8*half, col = nn
          int co = cobase + v + 8 * half;
          out[((size_t)(b * COUT_ + co) * HO_ + hpix) * WO_ + wpix] = acc[v];
        }
      }
    }
  }
}

// ---------------------------------------------------------------------------
extern "C" void kernel_launch(void* const* d_in, const int* in_sizes, int n_in,
                              void* d_out, int out_size, void* d_ws, size_t ws_size,
                              hipStream_t stream) {
  const float* x  = (const float*)d_in[0];
  const float* w1 = (const float*)d_in[1];
  const float* b1 = (const float*)d_in[2];
  const float* w2 = (const float*)d_in[3];
  const float* b2 = (const float*)d_in[4];
  const float* wg = (const float*)d_in[5];
  const float* bg = (const float*)d_in[6];
  float* out = (float*)d_out;

  char* ws = (char*)d_ws;
  float*         pooled = (float*)(ws);                   // 18 KB
  unsigned char* idxmap = (unsigned char*)(ws + 32768);   // 120 KB
  bf16_t*        kern   = (bf16_t*)(ws + 163840);         // 4.5 MB
  bf16_t*        xbfT   = (bf16_t*)(ws + 4882432);        // 16.3 MB

  pool_kernel<<<B_ * 9, 256, 0, stream>>>(x, pooled);
  kerngen_kernel<<<B_ * R_, 256, 0, stream>>>(pooled, w1, b1, w2, b2, kern);
  int ncvt = B_ * H_ * W_;
  cvt_kernel<<<(ncvt + 255) / 256, 256, 0, stream>>>(x, xbfT);
  int npix = B_ * HO_ * WO_;
  guide_kernel<<<(npix + 255) / 256, 256, 0, stream>>>(x, wg, bg, idxmap);
  drconv_main<<<B_ * HT_ * 8, 128, 0, stream>>>(xbfT, kern, idxmap, out);
}
